// Network_43731357008214
// MI455X (gfx1250) — matrix operand order repaired and checked by reading, not compiled, not run on hardware
//
#include <hip/hip_runtime.h>
#include <hip/hip_bf16.h>

// ---------------------------------------------------------------------------
// Problem constants
// ---------------------------------------------------------------------------
#define N_RAYS   4096
#define N_BOX    10
#define SALL     84            // N_FINE(64) + 2*N_BOX(20)
#define NPTS     (N_RAYS * SALL)   // 344064  (multiple of 256)
#define K0P      96            // 66 features padded to 96 (mult of 32)
#define HID      256
#define NOUT     65            // 50 (raw) + 15 (box3d)
#define NOUT_PAD 128           // padded WT rows for last GEMM
#define RAW_STRIDE 80          // f32 row stride of head output
#define DISTV    100.0f

// output offsets (floats), concat in reference return order
#define OFF_RGB   0                      // 4096*3
#define OFF_DEPTH 12288                  // 4096
#define OFF_ACC   16384                  // 4096
#define OFF_SEM   20480                  // 4096*46
#define OFF_FIX   208896                 // 4096*50
#define OFF_OH    413696                 // 344064*50
#define OFF_BOX   17616896               // 344064*15

typedef _Float16 v8h  __attribute__((ext_vector_type(8)));
typedef _Float16 v16h __attribute__((ext_vector_type(16)));
typedef float    v8f  __attribute__((ext_vector_type(8)));

// ---------------------------------------------------------------------------
// CDNA5 async global->LDS copy (probe confirmed builtin exists; param 0 is
// a global-AS pointer to int __attribute__((vector_size(16))))
// ---------------------------------------------------------------------------
#if __has_builtin(__builtin_amdgcn_global_load_async_to_lds_b128)
#define HAVE_ASYNC_LDS 1
typedef int v4i_ __attribute__((vector_size(16)));
typedef __attribute__((address_space(1))) v4i_* gptr_b128;
typedef __attribute__((address_space(3))) v4i_* lptr_b128;
#endif

__device__ __forceinline__ void stage16(const _Float16* src, _Float16* dst) {
#ifdef HAVE_ASYNC_LDS
    __builtin_amdgcn_global_load_async_to_lds_b128(
        (gptr_b128)(void*)const_cast<_Float16*>(src),
        (lptr_b128)(void*)dst, 0, 0);
#else
    *reinterpret_cast<v8h*>(dst) = *reinterpret_cast<const v8h*>(src);
#endif
}

__device__ __forceinline__ void stage_fence() {
#ifdef HAVE_ASYNC_LDS
#if __has_builtin(__builtin_amdgcn_s_wait_asynccnt)
    __builtin_amdgcn_s_wait_asynccnt(0);
#else
    asm volatile("s_wait_asynccnt 0" ::: "memory");
#endif
#endif
}

// ---------------------------------------------------------------------------
// 1) Weight prep: transpose + pad to f16, WT[n][k] layouts (K contiguous)
// ---------------------------------------------------------------------------
__global__ __launch_bounds__(256) void prep_weights(
    const float* __restrict__ W0, const float* __restrict__ W1,
    const float* __restrict__ Wo, const float* __restrict__ bo,
    const float* __restrict__ Wb, const float* __restrict__ bb,
    _Float16* __restrict__ WT0, _Float16* __restrict__ WT1,
    _Float16* __restrict__ WTO, float* __restrict__ biasO)
{
    int id = blockIdx.x * 256 + threadIdx.x;               // grid covers 256*256
    if (id < 256 * K0P) {                                  // WT0: [256][96]
        int n = id / K0P, k = id % K0P;
        WT0[id] = (_Float16)(k < 66 ? W0[k * 256 + n] : 0.0f);
    }
    if (id < 256 * 256) {                                  // WT1: [256][256]
        int n = id / 256, k = id % 256;
        WT1[id] = (_Float16)W1[k * 256 + n];
    }
    if (id < NOUT_PAD * 256) {                             // WTO: [128][256]
        int n = id / 256, k = id % 256;
        float v = 0.0f;
        if (n < 50)       v = Wo[k * 50 + n];
        else if (n < 65)  v = Wb[k * 15 + (n - 50)];
        WTO[id] = (_Float16)v;
    }
    if (id < NOUT_PAD) {
        float v = 0.0f;
        if (id < 50)      v = bo[id];
        else if (id < 65) v = bb[id - 50];
        biasO[id] = v;
    }
}

// ---------------------------------------------------------------------------
// 2) Per-ray z sampling (deterministic PDF resample + sort in LDS)
// ---------------------------------------------------------------------------
__global__ __launch_bounds__(64) void sample_kernel(
    const float* __restrict__ isect, float* __restrict__ Z)
{
    __shared__ float lds[64 * 128];
    int ray = blockIdx.x * 64 + threadIdx.x;
    float* buf = lds + threadIdx.x * 128;    // [0..39]=zc, [40..123]=z_vals

    const float* it = isect + (size_t)ray * (N_BOX * 4);
    float nearb[N_BOX], farb[N_BOX];
#pragma unroll
    for (int b = 0; b < N_BOX; ++b) { nearb[b] = it[b*4+0]; farb[b] = it[b*4+1]; }

    // coarse samples: 10 boxes x 4 linspace points
#pragma unroll
    for (int b = 0; b < N_BOX; ++b)
#pragma unroll
        for (int j = 0; j < 4; ++j)
            buf[b*4+j] = nearb[b] + (farb[b]-nearb[b]) * (float)j * (1.0f/3.0f);

    // sort 40 (insertion)
    for (int i = 1; i < 40; ++i) {
        float key = buf[i]; int j = i - 1;
        while (j >= 0 && buf[j] > key) { buf[j+1] = buf[j]; --j; }
        buf[j+1] = key;
    }

    // sample_pdf_det with uniform pdf over 39 weights -> cdf[k]=k/39
    for (int i = 0; i < 64; ++i) {
        float u = (float)i * (1.0f/63.0f);
        int cnt = 0;
        for (int k = 0; k < 40; ++k) cnt += (u >= (float)k * (1.0f/39.0f)) ? 1 : 0;
        int below = cnt - 1; if (below < 0) below = 0; if (below > 39) below = 39;
        int above = cnt;     if (above < 0) above = 0; if (above > 39) above = 39;
        float cb = (float)below * (1.0f/39.0f);
        float ca = (float)above * (1.0f/39.0f);
        float d  = ca - cb; if (d < 1e-5f) d = 1.0f;
        float t  = (u - cb) / d;
        float zb = buf[below], za = buf[above];
        buf[40 + i] = zb + t * (za - zb);
    }
    // bounds
#pragma unroll
    for (int b = 0; b < N_BOX; ++b) {
        buf[40 + 64 + b]      = nearb[b] - 1e-5f;
        buf[40 + 64 + 10 + b] = farb[b] + 1e-5f;
    }
    // background replacement for z<0 (hash PRNG stand-in for jax.random)
    for (int i = 0; i < SALL; ++i) {
        float z = buf[40 + i];
        if (z < 0.0f) {
            unsigned h = (unsigned)(ray * 2654435761u) ^ (unsigned)(i * 40503u + 0x9e3779b9u);
            h ^= h >> 16; h *= 0x85ebca6bu; h ^= h >> 13;
            buf[40 + i] = DISTV + 20.0f * ((float)(h & 0xFFFFFFu) * (1.0f/16777216.0f));
        }
    }
    // sort 84
    for (int i = 1; i < SALL; ++i) {
        float key = buf[40 + i]; int j = i - 1;
        while (j >= 0 && buf[40 + j] > key) { buf[40 + j + 1] = buf[40 + j]; --j; }
        buf[40 + j + 1] = key;
    }
    float* zo = Z + (size_t)ray * SALL;
    for (int i = 0; i < SALL; ++i) zo[i] = buf[40 + i];
}

// ---------------------------------------------------------------------------
// 3) Positional-encoded features -> f16 [NPTS][96]
// ---------------------------------------------------------------------------
__global__ __launch_bounds__(256) void feature_kernel(
    const float* __restrict__ rays, const float* __restrict__ Z,
    _Float16* __restrict__ X)
{
    int idx = blockIdx.x * 256 + threadIdx.x;    // < NPTS exactly
    int ray = idx / SALL;
    float z = Z[idx];
    const float* rp = rays + (size_t)ray * 6;
    float ox = rp[0], oy = rp[1], oz = rp[2];
    float dx = rp[3], dy = rp[4], dz = rp[5];
    float scale = sqrtf(dx*dx + dy*dy + dz*dz);
    float zs = z / scale;
    float px = (ox + dx * zs) * (1.0f / DISTV);
    float py = (oy + dy * zs) * (1.0f / DISTV);
    float pz = (oz + dz * zs) * (1.0f / DISTV);

    _Float16* xp = X + (size_t)idx * K0P;
    xp[0] = (_Float16)px; xp[1] = (_Float16)py; xp[2] = (_Float16)pz;
    float fx = px, fy = py, fz = pz;
#pragma unroll
    for (int l = 0; l < 10; ++l) {
        xp[3 + 6*l + 0] = (_Float16)__sinf(fx);
        xp[3 + 6*l + 1] = (_Float16)__sinf(fy);
        xp[3 + 6*l + 2] = (_Float16)__sinf(fz);
        xp[3 + 6*l + 3] = (_Float16)__cosf(fx);
        xp[3 + 6*l + 4] = (_Float16)__cosf(fy);
        xp[3 + 6*l + 5] = (_Float16)__cosf(fz);
        fx *= 2.0f; fy *= 2.0f; fz *= 2.0f;
    }
    xp[63] = (_Float16)dx; xp[64] = (_Float16)dy; xp[65] = (_Float16)dz;
#pragma unroll
    for (int k = 66; k < K0P; ++k) xp[k] = (_Float16)0.0f;
}

// ---------------------------------------------------------------------------
// 4) WMMA GEMM:  C[M][Cstride] = act(A[M][KP] * WT[n][KP]^T + bias)
//    Block: 256 thr = 8 waves; block tile 256x64; wave tile 32x64
//    (8x v_wmma_f32_16x16x32_f16 per K-step). B tile staged in LDS once.
// ---------------------------------------------------------------------------
template<int KP>
__global__ __launch_bounds__(256) void gemm_wmma(
    const _Float16* __restrict__ A, const _Float16* __restrict__ WT,
    const float* __restrict__ bias,
    _Float16* __restrict__ Ch, float* __restrict__ Cf,
    int Cstride, int Nout, int relu)
{
    constexpr int LSTR = KP + 8;                 // padded LDS row stride (halves)
    __shared__ _Float16 Bs[64 * LSTR];

    const int lane = threadIdx.x & 31;
    const int wave = threadIdx.x >> 5;
    const int row0 = blockIdx.x * 256 + wave * 32;
    const int col0 = blockIdx.y * 64;
    const int half = lane >> 4;                  // 0: lanes 0-15, 1: lanes 16-31
    const int l15  = lane & 15;

    // ---- stage B tile (64 rows x KP halves) into LDS, 16B chunks ----
    constexpr int CPR    = KP / 8;               // 16B chunks per row
    constexpr int CHUNKS = 64 * CPR;             // multiple of 256
#pragma unroll
    for (int c = threadIdx.x; c < CHUNKS; c += 256) {
        int r = c / CPR, i = c % CPR;
        stage16(WT + (size_t)(col0 + r) * KP + i * 8, &Bs[r * LSTR + i * 8]);
    }
    stage_fence();
    __syncthreads();

    const v8f vzero = {0.f,0.f,0.f,0.f,0.f,0.f,0.f,0.f};
    v8f acc[8];                                  // [mt*4 + nt]
#pragma unroll
    for (int i = 0; i < 8; ++i) acc[i] = vzero;

    const _Float16* arow0 = A + (size_t)(row0 + l15) * KP;
    const _Float16* arow1 = A + (size_t)(row0 + 16 + l15) * KP;

#pragma unroll
    for (int kt = 0; kt < KP; kt += 32) {
        // A fragments: 16-bit A 16x32 layout (ISA 7.12.2)
        union { v16h v; v8h h[2]; } a0, a1;
        a0.h[0] = *reinterpret_cast<const v8h*>(arow0 + kt + 8 * half);
        a0.h[1] = *reinterpret_cast<const v8h*>(arow0 + kt + 16 + 8 * half);
        a1.h[0] = *reinterpret_cast<const v8h*>(arow1 + kt + 8 * half);
        a1.h[1] = *reinterpret_cast<const v8h*>(arow1 + kt + 16 + 8 * half);
        // B fragments from LDS: lane holds col n, K = kt + 8*half + j (j<8), kt + 16 + 8*half + (j-8) (j>=8)
        v16h bf[4];
#pragma unroll
        for (int nt = 0; nt < 4; ++nt) {
            const _Float16* bp = &Bs[(nt * 16 + l15) * LSTR + kt + 8 * half];
            union { v16h v; v8h h[2]; } b;
            b.h[0] = *reinterpret_cast<const v8h*>(bp);
            b.h[1] = *reinterpret_cast<const v8h*>(bp + 16);
            bf[nt] = b.v;
        }
#pragma unroll
        for (int nt = 0; nt < 4; ++nt) {
            acc[nt]     = __builtin_amdgcn_wmma_f32_16x16x32_f16(
                false, a0.v, false, bf[nt], (short)0, acc[nt],     false, false);
            acc[4 + nt] = __builtin_amdgcn_wmma_f32_16x16x32_f16(
                false, a1.v, false, bf[nt], (short)0, acc[4 + nt], false, false);
        }
    }

    // epilogue: C/D layout — lane: n = l15 (+tile), VGPR r: m = r + 8*half
#pragma unroll
    for (int mt = 0; mt < 2; ++mt)
#pragma unroll
    for (int nt = 0; nt < 4; ++nt) {
        int n = col0 + nt * 16 + l15;
        if (n >= Nout) continue;
        float bv = bias[n];
#pragma unroll
        for (int r = 0; r < 8; ++r) {
            int m = row0 + mt * 16 + r + 8 * half;
            float v = acc[mt * 4 + nt][r] + bv;
            if (relu) v = fmaxf(v, 0.0f);
            if (Ch) Ch[(size_t)m * Cstride + n] = (_Float16)v;
            else    Cf[(size_t)m * Cstride + n] = v;
        }
    }
}

// ---------------------------------------------------------------------------
// 5) Volumetric rendering + one-hot semantics + all outputs
// ---------------------------------------------------------------------------
__global__ __launch_bounds__(64) void render_kernel(
    const float* __restrict__ rays, const float* __restrict__ isect,
    const float* __restrict__ Z, const float* __restrict__ RAW,
    float* __restrict__ out)
{
    int ray = blockIdx.x * 64 + threadIdx.x;
    const float* rp = rays + (size_t)ray * 6;
    float dx = rp[3], dy = rp[4], dz = rp[5];
    float scale = sqrtf(dx*dx + dy*dy + dz*dz);

    const float* it = isect + (size_t)ray * (N_BOX * 4);
    float nearb[N_BOX], farb[N_BOX]; int lab[N_BOX];
#pragma unroll
    for (int b = 0; b < N_BOX; ++b) {
        nearb[b] = it[b*4+0]; farb[b] = it[b*4+1]; lab[b] = (int)it[b*4+3];
    }

    float T = 1.0f, cr = 0.f, cg = 0.f, cb = 0.f, depth = 0.f, accm = 0.f;
    float sem[46], fix[50];
#pragma unroll
    for (int c = 0; c < 46; ++c) sem[c] = 0.f;
#pragma unroll
    for (int c = 0; c < 50; ++c) fix[c] = 0.f;

    const float* Zr = Z + (size_t)ray * SALL;
    float* OH  = out + OFF_OH;
    float* BOX = out + OFF_BOX;

    for (int s = 0; s < SALL; ++s) {
        float z  = Zr[s];
        float zr = z / scale;
        float dist = (s < SALL - 1) ? (Zr[s + 1] - z) : 1e10f * scale;
        const float* raw = RAW + ((size_t)ray * SALL + s) * RAW_STRIDE;

        unsigned long long oh = 0ull;
        bool bound = false;
#pragma unroll
        for (int b = 0; b < N_BOX; ++b) {
            if (z > nearb[b] && z < farb[b]) oh |= 1ull << (lab[b] & 63);
            float dfar  = z - farb[b];
            float dnear = nearb[b] - z;
            bound = bound || (dfar < 0.001f && dfar > 0.f) || (dnear > 0.f && dnear < 0.001f);
        }
        bool bbox = (z < DISTV) && (oh == 0ull);
        if (bbox) oh |= 1ull;                                   // class 0
        if ((z > DISTV) && (oh == 0ull)) oh |= 1ull << 23;      // class 23
        // merges: 39->41, {27..31}->26, 9->8, 35->13
        if (oh & (1ull << 39)) { oh |= 1ull << 41; oh &= ~(1ull << 39); }
        {
            const unsigned long long carm =
                (1ull<<27)|(1ull<<28)|(1ull<<29)|(1ull<<30)|(1ull<<31);
            if (oh & carm) { oh |= 1ull << 26; oh &= ~carm; }
        }
        if (oh & (1ull << 9))  { oh |= 1ull << 8;  oh &= ~(1ull << 9);  }
        if (oh & (1ull << 35)) { oh |= 1ull << 13; oh &= ~(1ull << 35); }

        float sigma = raw[3];
        if (bbox || bound) sigma = 0.0f;
        float alpha = 1.0f - __expf(-fmaxf(sigma, 0.0f) * dist);
        float w = alpha * T;
        T *= (1.0f - alpha + 1e-10f);

        cr += w * (1.0f / (1.0f + __expf(-raw[0])));
        cg += w * (1.0f / (1.0f + __expf(-raw[1])));
        cb += w * (1.0f / (1.0f + __expf(-raw[2])));
        depth += w * zr;
        accm  += w;
#pragma unroll
        for (int c = 0; c < 46; ++c) sem[c] += w * raw[4 + c];
#pragma unroll
        for (int c = 0; c < 50; ++c) {
            float o = (float)((oh >> c) & 1ull);
            fix[c] += w * o;
            OH[((size_t)ray * SALL + s) * 50 + c] = o;
        }
#pragma unroll
        for (int j = 0; j < 15; ++j)
            BOX[((size_t)ray * SALL + s) * 15 + j] = raw[50 + j];
    }

    out[OFF_RGB + ray * 3 + 0] = cr;
    out[OFF_RGB + ray * 3 + 1] = cg;
    out[OFF_RGB + ray * 3 + 2] = cb;
    out[OFF_DEPTH + ray] = depth;
    out[OFF_ACC   + ray] = accm;
#pragma unroll
    for (int c = 0; c < 46; ++c) out[OFF_SEM + ray * 46 + c] = sem[c];
#pragma unroll
    for (int c = 0; c < 50; ++c) out[OFF_FIX + ray * 50 + c] = fix[c];
}

// ---------------------------------------------------------------------------
// Launch
// ---------------------------------------------------------------------------
extern "C" void kernel_launch(void* const* d_in, const int* in_sizes, int n_in,
                              void* d_out, int out_size, void* d_ws, size_t ws_size,
                              hipStream_t stream)
{
    const float* rays  = (const float*)d_in[0];
    const float* isect = (const float*)d_in[1];
    const float* W0    = (const float*)d_in[2];
    const float* b0    = (const float*)d_in[3];
    const float* W1    = (const float*)d_in[4];
    const float* b1    = (const float*)d_in[5];
    const float* Wo    = (const float*)d_in[6];
    const float* bo    = (const float*)d_in[7];
    const float* Wb    = (const float*)d_in[8];
    const float* bb    = (const float*)d_in[9];
    float* out = (float*)d_out;
    char* ws = (char*)d_ws;

    const size_t M = NPTS;
    size_t off = 0;
    auto take = [&](size_t bytes) {
        size_t o = off;
        off = (off + bytes + 255) & ~(size_t)255;
        return o;
    };
    float*    Z     = (float*)   (ws + take(M * sizeof(float)));
    _Float16* X     = (_Float16*)(ws + take(M * K0P * sizeof(_Float16)));
    _Float16* H0    = (_Float16*)(ws + take(M * HID * sizeof(_Float16)));
    _Float16* H1    = (_Float16*)(ws + take(M * HID * sizeof(_Float16)));
    float*    RAW   = (float*)   (ws + take(M * RAW_STRIDE * sizeof(float)));
    _Float16* WT0   = (_Float16*)(ws + take(256 * K0P * sizeof(_Float16)));
    _Float16* WT1   = (_Float16*)(ws + take(256 * 256 * sizeof(_Float16)));
    _Float16* WTO   = (_Float16*)(ws + take(NOUT_PAD * 256 * sizeof(_Float16)));
    float*    biasO = (float*)   (ws + take(NOUT_PAD * sizeof(float)));
    (void)ws_size; (void)in_sizes; (void)n_in; (void)out_size;

    prep_weights<<<dim3(256), dim3(256), 0, stream>>>(
        W0, W1, Wo, bo, Wb, bb, WT0, WT1, WTO, biasO);

    sample_kernel<<<dim3(N_RAYS / 64), dim3(64), 0, stream>>>(isect, Z);

    feature_kernel<<<dim3(NPTS / 256), dim3(256), 0, stream>>>(rays, Z, X);

    // layer 0: X[M,96] x WT0 -> H0[M,256] (relu, f16)
    gemm_wmma<K0P><<<dim3(M / 256, 4), dim3(256), 0, stream>>>(
        X, WT0, b0, H0, nullptr, HID, HID, 1);
    // layer 1: H0 x WT1 -> H1[M,256] (relu, f16)
    gemm_wmma<HID><<<dim3(M / 256, 4), dim3(256), 0, stream>>>(
        H0, WT1, b1, H1, nullptr, HID, HID, 1);
    // head: H1 x WTO -> RAW[M,80] (linear, f32; cols 0..49 raw, 50..64 box3d)
    gemm_wmma<HID><<<dim3(M / 256, 2), dim3(256), 0, stream>>>(
        H1, WTO, biasO, nullptr, RAW, RAW_STRIDE, NOUT, 0);

    render_kernel<<<dim3(N_RAYS / 64), dim3(64), 0, stream>>>(
        rays, isect, Z, RAW, out);
}
